// Attention_11364483465611
// MI455X (gfx1250) — matrix operand order repaired and checked by reading, not compiled, not run on hardware
//
#include <hip/hip_runtime.h>
#include <hip/hip_bf16.h>

// ---------------------------------------------------------------------------
// Gemma-2-style attention for MI455X (gfx1250): bf16 WMMA everywhere.
//   K1: q/k/v projections + RoPE (+ q scale)    -> bf16 buffers in d_ws
//   K2: flash attention, window=1024, softcap50 -> enc bf16 in d_ws
//       (K tile staged to LDS via Tensor Data Mover when available)
//   K3: output projection                        -> f32 d_out
// All LDS tiles use padded row strides (odd multiple of 4 dwords, 16B-aligned)
// so 16-lane fragment reads hit 16 distinct banks and still merge to b128.
// ---------------------------------------------------------------------------

#define T_SEQ  2048
#define DMODEL 2048
#define HD     128
#define NHEAD  16
#define NKV    4
#define WINDOW 1024

// padded LDS row strides (in u16 elements)
#define GEMM_STRIDE 40    // 32 cols + 8 pad   (80B rows, 16B aligned, 20 u32)
#define KT_STRIDE   136   // 128 cols + 8 pad  (272B rows, 16B aligned, 68 u32)
#define VT_STRIDE   72    // 64 cols + 8 pad   (144B rows, 16B aligned, 36 u32)
#define P_STRIDE    72    // 64 cols + 8 pad
#define P_WAVE      (16 * P_STRIDE)

#if defined(__has_builtin)
# if __has_builtin(__builtin_amdgcn_tensor_load_to_lds) && __has_builtin(__builtin_amdgcn_s_wait_tensorcnt)
#  define USE_TDM 1
# endif
#endif
#ifndef USE_TDM
# define USE_TDM 0
#endif

typedef __attribute__((ext_vector_type(16))) __bf16    v16bf;
typedef __attribute__((ext_vector_type(8)))  float     v8f;
typedef __attribute__((ext_vector_type(4)))  unsigned  v4u;
typedef __attribute__((ext_vector_type(8)))  int       v8i;
typedef __attribute__((ext_vector_type(4)))  int       v4i;

union FragBF {
    unsigned u[8];
    v16bf    v;
};

__device__ inline unsigned short f2bf(float f) {
    // round-to-nearest-even f32 -> bf16 (bit pattern)
    unsigned u = __float_as_uint(f);
    unsigned r = u + 0x7fffu + ((u >> 16) & 1u);
    return (unsigned short)(r >> 16);
}

__device__ inline v8f wmma_bf16(v16bf a, v16bf b, v8f c) {
    return __builtin_amdgcn_wmma_f32_16x16x32_bf16(
        /*neg_a=*/false, a, /*neg_b=*/false, b,
        /*c_mod=*/(short)0, c, /*reuse_a=*/false, /*reuse_b=*/false);
}

// A-fragment (16x32 bf16) K index for vgpr i, lane-half hi (ISA 7.12.2)
__device__ inline int a_kidx(int i, int hi) {
    return ((i >> 2) << 4) + hi * 8 + ((i & 3) << 1);
}
// B-fragment (32x16 bf16) K index for vgpr i, lane-half hi
__device__ inline int b_kidx(int i, int hi) {
    return ((i >> 2) << 4) + hi * 8 + ((i & 3) << 1);
}

#if USE_TDM
// TDM: load a 64(rows) x 128(bf16) tile (row stride 128 elems in memory) into
// LDS at lds_off, hardware-padding +4 DWORDs after every 64 DWORDs so the LDS
// row stride becomes 272B (= KT_STRIDE u16). Wave-level op; EXEC ignored.
// 6-arg builtin form (amdgpu-toolchain / clang-23):
//   (uint32x4 g0, int32x8 g1, int32x4 g2, int32x4 g3, int32x8 g4, i32 cpol)
__device__ inline void tdm_load_k_tile(const unsigned short* gptr, unsigned lds_off) {
    unsigned long long ga = (unsigned long long)(uintptr_t)gptr;
    v4u g0;
    g0[0] = 1u;                                               // count=1, user D#
    g0[1] = lds_off;                                          // D#.lds_addr
    g0[2] = (unsigned)(ga & 0xffffffffu);                     // global_addr[31:0]
    g0[3] = (unsigned)((ga >> 32) & 0x01ffffffu) | (2u << 30);// [56:32] | type=2
    v8i g1;
    g1[0] = (int)0x07510000u;   // data_size=2B | pad_enable | interval=64dw | amount=4dw
    g1[1] = (int)(128u << 16);  // tensor_dim0 = 128 (low16 at bits 63:48)
    g1[2] = (int)(2048u << 16); // tensor_dim0 hi=0 | tensor_dim1 low16 = 2048
    g1[3] = (int)(128u << 16);  // tensor_dim1 hi=0 | tile_dim0 = 128
    g1[4] = 64;                 // tile_dim1 = 64, tile_dim2 = 0
    g1[5] = 128;                // tensor_dim0_stride = 128
    g1[6] = 0;
    g1[7] = 0;
    v4i gz4 = {0, 0, 0, 0};
    v8i gz8 = {0, 0, 0, 0, 0, 0, 0, 0};
    __builtin_amdgcn_tensor_load_to_lds(g0, g1, gz4, gz4, gz8, 0);
}
#endif

// ---------------------------------------------------------------------------
// Kernel 1: QKV projection + RoPE.
// grid = (T/128, 24, B): y in [0,16) -> Q head, [16,20) -> K head, [20,24) -> V head
// block = 256 (8 waves); each wave owns a 16x128 output strip.
// ---------------------------------------------------------------------------
__global__ __launch_bounds__(256) void qkv_rope_kernel(
    const float* __restrict__ x, const int* __restrict__ segpos,
    const float* __restrict__ wq, const float* __restrict__ wkv,
    unsigned short* __restrict__ qb, unsigned short* __restrict__ kb,
    unsigned short* __restrict__ vb)
{
    __shared__ unsigned short lds_a[128 * GEMM_STRIDE];  // [row][k]        bf16
    __shared__ unsigned short lds_b[128 * GEMM_STRIDE];  // [col][k] (K-T)  bf16

    const int t0  = blockIdx.x * 128;
    const int hy  = blockIdx.y;
    const int bz  = blockIdx.z;
    const int tid = threadIdx.x;
    const int w   = tid >> 5;
    const int l   = tid & 31;
    const int l16 = l & 15;
    const int hi  = l >> 4;

    const float* W;
    if (hy < NHEAD)            W = wq  + (size_t)hy * DMODEL * HD;
    else if (hy < NHEAD + NKV) W = wkv + (size_t)(hy - NHEAD) * DMODEL * HD;               // K (c=0)
    else                       W = wkv + (size_t)(NKV + (hy - NHEAD - NKV)) * DMODEL * HD; // V (c=1)

    v8f acc[8];
#pragma unroll
    for (int n = 0; n < 8; ++n)
#pragma unroll
        for (int j = 0; j < 8; ++j) acc[n][j] = 0.0f;

    for (int kb0 = 0; kb0 < DMODEL; kb0 += 32) {
        __syncthreads();
        // prefetch next K-step tiles while this step computes
        if (kb0 + 32 < DMODEL) {
            __builtin_prefetch(&x[(size_t)(bz * T_SEQ + t0 + (tid >> 5) * 16) * DMODEL + kb0 + 32 + (tid & 31)], 0, 3);
            __builtin_prefetch(&W[(size_t)(kb0 + 32 + (tid >> 3)) * HD + (tid & 7) * 16], 0, 3);
        }
        // stage A (x rows, f32->bf16), row-major in LDS
#pragma unroll
        for (int it = 0; it < 16; ++it) {
            int idx = tid + it * 256;
            int r = idx >> 5, c = idx & 31;
            float f = x[(size_t)(bz * T_SEQ + t0 + r) * DMODEL + kb0 + c];
            lds_a[r * GEMM_STRIDE + c] = f2bf(f);
        }
        // stage B (W rows kb0..+31), store K-transposed: lds_b[col][k]
#pragma unroll
        for (int it = 0; it < 16; ++it) {
            int idx = tid + it * 256;
            int kr = idx >> 7, c = idx & 127;
            float f = W[(size_t)(kb0 + kr) * HD + c];
            lds_b[c * GEMM_STRIDE + kr] = f2bf(f);
        }
        __syncthreads();

        const unsigned* a32 = (const unsigned*)lds_a;
        const unsigned* b32 = (const unsigned*)lds_b;
        FragBF af;
        const int arow = w * 16 + l16;
#pragma unroll
        for (int i = 0; i < 8; ++i)
            af.u[i] = a32[(arow * GEMM_STRIDE + a_kidx(i, hi)) >> 1];

#pragma unroll
        for (int n = 0; n < 8; ++n) {
            FragBF bf;
            const int col = n * 16 + l16;
#pragma unroll
            for (int i = 0; i < 8; ++i)
                bf.u[i] = b32[(col * GEMM_STRIDE + b_kidx(i, hi)) >> 1];
            acc[n] = wmma_bf16(af.v, bf.v, acc[n]);
        }
    }

    // RoPE (Q and K heads): pair (col h, col h+64) = (subtile n, subtile n+4), same lane.
    if (hy < NHEAD + NKV) {
#pragma unroll
        for (int r = 0; r < 8; ++r) {
            const int t = t0 + w * 16 + r + 8 * hi;
            const float p = (float)segpos[bz * T_SEQ + t];
#pragma unroll
            for (int n = 0; n < 4; ++n) {
                const int hrot = n * 16 + l16;                 // 0..63
                float inv_ts = __powf(10000.0f, -(float)hrot * (1.0f / 64.0f));
                float ang = p * inv_ts;
                float s = __sinf(ang), c = __cosf(ang);
                float fr  = acc[n][r];
                float sec = acc[n + 4][r];
                float nf = fr * c - sec * s;
                float ns = sec * c + fr * s;
                if (hy < NHEAD) { nf *= 0.08838834764831845f; ns *= 0.08838834764831845f; }
                acc[n][r]     = nf;
                acc[n + 4][r] = ns;
            }
        }
    }

    unsigned short* dst;
    size_t base;
    if (hy < NHEAD)            { dst = qb; base = (size_t)(bz * NHEAD + hy) * T_SEQ * HD; }
    else if (hy < NHEAD + NKV) { dst = kb; base = (size_t)(bz * NKV + (hy - NHEAD)) * T_SEQ * HD; }
    else                       { dst = vb; base = (size_t)(bz * NKV + (hy - NHEAD - NKV)) * T_SEQ * HD; }
#pragma unroll
    for (int r = 0; r < 8; ++r) {
        const int t = t0 + w * 16 + r + 8 * hi;
#pragma unroll
        for (int n = 0; n < 8; ++n)
            dst[base + (size_t)t * HD + n * 16 + l16] = f2bf(acc[n][r]);
    }
}

// ---------------------------------------------------------------------------
// Kernel 2: flash attention (window 1024, causal, softcap 50).
// grid = (T/64, NHEAD, B), block = 128 (4 waves); wave owns 16 q rows.
// K block staged to LDS once per block (TDM when available), V^T staged
// cooperatively (needs a transpose TDM cannot do).
// ---------------------------------------------------------------------------
__global__ __launch_bounds__(128) void attn_kernel(
    const unsigned short* __restrict__ qbuf, const unsigned short* __restrict__ kbuf,
    const unsigned short* __restrict__ vbuf, unsigned short* __restrict__ enc)
{
    __shared__ unsigned short kt[64 * KT_STRIDE];   // K block: [s][h], bf16, padded rows
    __shared__ unsigned short vt[HD * VT_STRIDE];   // V^T:     [h][s], bf16, padded rows
    __shared__ unsigned short plds[4 * P_WAVE];     // per-wave P tile, padded rows

    const int t0  = blockIdx.x * 64;
    const int h   = blockIdx.y;
    const int bz  = blockIdx.z;
    const int kv  = h >> 2;                        // GQA group
    const int tid = threadIdx.x;
    const int w   = tid >> 5;
    const int l   = tid & 31;
    const int l16 = l & 15;
    const int hi  = l >> 4;
    const int tq0 = t0 + w * 16;

    const unsigned* q32 = (const unsigned*)qbuf;
    const unsigned* v32 = (const unsigned*)vbuf;
    const size_t qbase = (size_t)(bz * NHEAD + h) * T_SEQ * HD;
    const size_t kbase = (size_t)(bz * NKV + kv) * T_SEQ * HD;
    const size_t vbase = (size_t)(bz * NKV + kv) * T_SEQ * HD;

    // Q A-fragments (16 rows x 128 K) straight from global bf16
    FragBF aq[4];
#pragma unroll
    for (int f = 0; f < 4; ++f)
#pragma unroll
        for (int i = 0; i < 8; ++i) {
            int k = f * 32 + a_kidx(i, hi);
            aq[f].u[i] = q32[(qbase + (size_t)(tq0 + l16) * HD + k) >> 1];
        }

    v8f oacc[8];
    float m[8], lsum[8];
#pragma unroll
    for (int n = 0; n < 8; ++n)
#pragma unroll
        for (int j = 0; j < 8; ++j) oacc[n][j] = 0.0f;
#pragma unroll
    for (int r = 0; r < 8; ++r) { m[r] = -1.0e4f; lsum[r] = 0.0f; }

#if USE_TDM
    const unsigned kt_off = (unsigned)(uintptr_t)(void*)kt;  // LDS byte offset
#endif

    const int s_begin = (t0 - WINDOW) > 0 ? (t0 - WINDOW) : 0;  // 64-aligned block start
    for (int sb = s_begin; sb < t0 + 64; sb += 64) {
        __syncthreads();
        // ---- stage K block [sb..sb+63][0..127] into LDS (padded rows) ----
#if USE_TDM
        if (w == 0)
            tdm_load_k_tile(kbuf + kbase + (size_t)sb * HD, kt_off);
#else
        {
            const unsigned* k32g = (const unsigned*)kbuf;
#pragma unroll
            for (int it = 0; it < 32; ++it) {
                int idx = tid + it * 128;
                int s = idx >> 6, hp = idx & 63;
                unsigned d = k32g[((kbase + (size_t)(sb + s) * HD) >> 1) + hp];
                *(unsigned*)&kt[s * KT_STRIDE + 2 * hp] = d;
            }
        }
#endif
        // ---- stage V^T into LDS (u32 = pair of h's per key) ----
#pragma unroll
        for (int it = 0; it < 32; ++it) {
            int idx = tid + it * 128;
            int s = idx >> 6, hp = idx & 63;
            unsigned d = v32[((vbase + (size_t)(sb + s) * HD) >> 1) + hp];
            vt[(2 * hp) * VT_STRIDE + s]     = (unsigned short)(d & 0xffffu);
            vt[(2 * hp + 1) * VT_STRIDE + s] = (unsigned short)(d >> 16);
        }
#if USE_TDM
        if (w == 0) __builtin_amdgcn_s_wait_tensorcnt(0);
#endif
        __syncthreads();

        // ---- S = Q K^T (4 subtiles of 16 keys), K fragments from LDS ----
        const unsigned* kt32 = (const unsigned*)kt;
        v8f sacc[4];
#pragma unroll
        for (int n = 0; n < 4; ++n)
#pragma unroll
            for (int j = 0; j < 8; ++j) sacc[n][j] = 0.0f;
#pragma unroll
        for (int n = 0; n < 4; ++n) {
            const int srel = n * 16 + l16;
#pragma unroll
            for (int f = 0; f < 4; ++f) {
                FragBF bk;
#pragma unroll
                for (int i = 0; i < 8; ++i) {
                    int kk = f * 32 + b_kidx(i, hi);
                    bk.u[i] = kt32[(srel * KT_STRIDE + kk) >> 1];
                }
                sacc[n] = wmma_bf16(aq[f].v, bk.v, sacc[n]);
            }
        }

        // ---- softcap + window/causal mask + row max (16-lane butterfly) ----
        float rmax[8];
#pragma unroll
        for (int r = 0; r < 8; ++r) {
            const int tr = tq0 + r + 8 * hi;
            float mx = -1.0e4f;
#pragma unroll
            for (int n = 0; n < 4; ++n) {
                float val = sacc[n][r];
                val = 50.0f * tanhf(val * 0.02f);
                int sc = sb + n * 16 + l16;
                bool ok = (sc <= tr) && (sc > tr - WINDOW);
                val = ok ? val : -1.0e4f;
                sacc[n][r] = val;
                mx = fmaxf(mx, val);
            }
#pragma unroll
            for (int d = 1; d < 16; d <<= 1) mx = fmaxf(mx, __shfl_xor(mx, d, 32));
            rmax[r] = mx;
        }

        // ---- online softmax update ----
#pragma unroll
        for (int r = 0; r < 8; ++r) {
            float mnew  = fmaxf(m[r], rmax[r]);
            float alpha = __expf(m[r] - mnew);
            m[r] = mnew;
            float rsum = 0.0f;
#pragma unroll
            for (int n = 0; n < 4; ++n) {
                float p = __expf(sacc[n][r] - mnew);
                sacc[n][r] = p;
                rsum += p;
            }
#pragma unroll
            for (int d = 1; d < 16; d <<= 1) rsum += __shfl_xor(rsum, d, 32);
            lsum[r] = lsum[r] * alpha + rsum;
#pragma unroll
            for (int n = 0; n < 8; ++n) oacc[n][r] *= alpha;
        }

        // ---- C-layout -> A-layout for P via per-wave LDS ----
#pragma unroll
        for (int r = 0; r < 8; ++r) {
            const int mrow = r + 8 * hi;
#pragma unroll
            for (int n = 0; n < 4; ++n)
                plds[w * P_WAVE + mrow * P_STRIDE + n * 16 + l16] = f2bf(sacc[n][r]);
        }
        __syncthreads();

        // ---- O += P V ----
        const unsigned* p32  = (const unsigned*)plds;
        const unsigned* vt32 = (const unsigned*)vt;
#pragma unroll
        for (int f = 0; f < 2; ++f) {
            FragBF ap;
#pragma unroll
            for (int i = 0; i < 8; ++i) {
                int k = f * 32 + a_kidx(i, hi);
                ap.u[i] = p32[(w * P_WAVE + l16 * P_STRIDE + k) >> 1];
            }
#pragma unroll
            for (int n = 0; n < 8; ++n) {
                FragBF bv;
                const int hcol = n * 16 + l16;
#pragma unroll
                for (int i = 0; i < 8; ++i) {
                    int kk = f * 32 + b_kidx(i, hi);
                    bv.u[i] = vt32[(hcol * VT_STRIDE + kk) >> 1];
                }
                oacc[n] = wmma_bf16(ap.v, bv.v, oacc[n]);
            }
        }
    }

    // ---- normalize + store enc[b][t][head][h] (bf16) ----
#pragma unroll
    for (int r = 0; r < 8; ++r) {
        const int tr = tq0 + r + 8 * hi;
        const float inv = 1.0f / lsum[r];
#pragma unroll
        for (int n = 0; n < 8; ++n)
            enc[((size_t)(bz * T_SEQ + tr) * NHEAD + h) * HD + n * 16 + l16] =
                f2bf(oacc[n][r] * inv);
    }
}

// ---------------------------------------------------------------------------
// Kernel 3: output projection  out[b,t,:] = enc[b,t,:2048] @ wvec[2048,2048]
// grid = (T/128, D/128, B), block = 256 (8 waves).
// ---------------------------------------------------------------------------
__global__ __launch_bounds__(256) void outproj_kernel(
    const unsigned short* __restrict__ enc, const float* __restrict__ wvec,
    float* __restrict__ out)
{
    __shared__ unsigned short lds_b[128 * GEMM_STRIDE];  // [col d][k], bf16

    const int t0  = blockIdx.x * 128;
    const int d0  = blockIdx.y * 128;
    const int bz  = blockIdx.z;
    const int tid = threadIdx.x;
    const int w   = tid >> 5;
    const int l   = tid & 31;
    const int l16 = l & 15;
    const int hi  = l >> 4;

    v8f acc[8];
#pragma unroll
    for (int n = 0; n < 8; ++n)
#pragma unroll
        for (int j = 0; j < 8; ++j) acc[n][j] = 0.0f;

    const unsigned* a32 = (const unsigned*)enc;
    const size_t abase = (size_t)bz * T_SEQ * DMODEL;

    for (int kb0 = 0; kb0 < DMODEL; kb0 += 32) {
        __syncthreads();
        if (kb0 + 32 < DMODEL)
            __builtin_prefetch(&wvec[(size_t)(kb0 + 32 + (tid >> 3)) * DMODEL + d0 + (tid & 7) * 16], 0, 3);
#pragma unroll
        for (int it = 0; it < 16; ++it) {
            int idx = tid + it * 256;
            int kr = idx >> 7, c = idx & 127;
            lds_b[c * GEMM_STRIDE + kr] = f2bf(wvec[(size_t)(kb0 + kr) * DMODEL + d0 + c]);
        }
        __syncthreads();

        FragBF af;
        const int arow = t0 + w * 16 + l16;
#pragma unroll
        for (int i = 0; i < 8; ++i) {
            int k = kb0 + a_kidx(i, hi);
            af.u[i] = a32[(abase + (size_t)arow * DMODEL + k) >> 1];
        }
        const unsigned* b32 = (const unsigned*)lds_b;
#pragma unroll
        for (int n = 0; n < 8; ++n) {
            FragBF bf;
            const int col = n * 16 + l16;
#pragma unroll
            for (int i = 0; i < 8; ++i)
                bf.u[i] = b32[(col * GEMM_STRIDE + b_kidx(i, hi)) >> 1];
            acc[n] = wmma_bf16(af.v, bf.v, acc[n]);
        }
    }

#pragma unroll
    for (int r = 0; r < 8; ++r) {
        const int t = t0 + w * 16 + r + 8 * hi;
#pragma unroll
        for (int n = 0; n < 8; ++n)
            out[(size_t)(bz * T_SEQ + t) * DMODEL + d0 + n * 16 + l16] = acc[n][r];
    }
}

// ---------------------------------------------------------------------------
extern "C" void kernel_launch(void* const* d_in, const int* in_sizes, int n_in,
                              void* d_out, int out_size, void* d_ws, size_t ws_size,
                              hipStream_t stream) {
    const float* x      = (const float*)d_in[0];
    const int*   segpos = (const int*)d_in[1];
    // d_in[2] = attn_mask (bool): mask is computed analytically (causal+window)
    const float* wq     = (const float*)d_in[3];
    const float* wkv    = (const float*)d_in[4];
    const float* wvec   = (const float*)d_in[5];
    float* out = (float*)d_out;

    // workspace layout (bf16 as u16): q | k | v | enc  (~40 MB total)
    unsigned short* qb   = (unsigned short*)d_ws;
    unsigned short* kbuf = qb   + (size_t)2 * NHEAD * T_SEQ * HD;
    unsigned short* vbuf = kbuf + (size_t)2 * NKV   * T_SEQ * HD;
    unsigned short* enc  = vbuf + (size_t)2 * NKV   * T_SEQ * HD;

    dim3 g1(T_SEQ / 128, NHEAD + 2 * NKV, 2);
    qkv_rope_kernel<<<g1, 256, 0, stream>>>(x, segpos, wq, wkv, qb, kbuf, vbuf);

    dim3 g2(T_SEQ / 64, NHEAD, 2);
    attn_kernel<<<g2, 128, 0, stream>>>(qb, kbuf, vbuf, enc);

    dim3 g3(T_SEQ / 128, DMODEL / 128, 2);
    outproj_kernel<<<g3, 256, 0, stream>>>(enc, wvec, out);
}
